// AdaptiveFrequencySelector_47931835023717
// MI455X (gfx1250) — matrix operand order repaired and checked by reading, not compiled, not run on hardware
//
#include <hip/hip_runtime.h>
#include <math.h>

#define DIM 512
#define NHEADS 8
#define HEAD_DIM 64
#define BATCH 8
#define FBINS 8192
#define MROWS 32   // rows per workgroup = two 16-row M-tiles sharing B-fragments

typedef __bf16 bf16_t;
typedef __attribute__((ext_vector_type(16))) __bf16 v16bf;
typedef __attribute__((ext_vector_type(8)))  __bf16 v8bf;
typedef __attribute__((ext_vector_type(8)))  float  v8f;
typedef __attribute__((ext_vector_type(4)))  float  v4f;

// ---- packed bf16 weight pool, element offsets (each 32x16 tile = 512 bf16) ----
#define PK_IMP_W1 0          // K=512  N=1024 : 16*64 tiles
#define PK_IMP_W2 524288     // K=1024 N=512  : 32*32 tiles
#define PK_IMP_W3 1048576    // K=512  Npad=16: 16 tiles
#define PK_B0_W1  1056768    // K=512  N=512
#define PK_B1_W1  1318912    // K=512  N=256
#define PK_B2_W1  1449984    // K=512  N=128
#define PK_B3_W1  1515520    // K=512  N=64
#define PK_B0_W2  1548288    // K=512  Npad=16
#define PK_B1_W2  1556480    // K=256
#define PK_B2_W2  1560576    // K=128
#define PK_B3_W2  1562624    // K=64
#define PK_QKM    1563648    // per-batch K=512 Npad=16 (8 * 16 tiles)
#define PK_TOTAL  1629184

// ---- workspace byte offsets ----
#define WS_MAGMEAN (PK_TOTAL * 2)
#define WS_KM      (WS_MAGMEAN + BATCH * DIM * 4)
#define WS_QBKM    (WS_KM + BATCH * DIM * 4)
#define WS_FINAL   (WS_QBKM + 256)

__device__ __forceinline__ float gelu_exact(float x) {
  return 0.5f * x * (1.0f + erff(x * 0.70710678118654752f));
}

// Build A-fragment (16x32 bf16, M x K) from row-major LDS tile.
// lanes 0-15: row=lane, halves[0..15]=K k0+0..15
// lanes16-31: row=lane-16, K k0+16..31
__device__ __forceinline__ v16bf load_a_frag(const bf16_t* base, int stride, int lane, int k0) {
  const int r  = lane & 15;
  const int ko = (lane & 16) ? 16 : 0;
  const bf16_t* p = base + r * stride + k0 + ko;
  v8bf lo = *(const v8bf*)(p);
  v8bf hi = *(const v8bf*)(p + 8);
  return __builtin_shufflevector(lo, hi, 0,1,2,3,4,5,6,7,8,9,10,11,12,13,14,15);
}

// two 16x16 M-tiles sharing each B-fragment: C0,C1 += A0,A1 * B over KT k-steps
__device__ __forceinline__ void gemm_tile2(const bf16_t* abase, int astride,
                                           const bf16_t* wbase, int KT, int lane,
                                           v8f& acc0, v8f& acc1) {
  const bf16_t* abase1 = abase + 16 * astride;
  for (int kt = 0; kt < KT; ++kt) {
    if (kt + 4 < KT)
      __builtin_prefetch(wbase + (size_t)(kt + 4) * 512 + lane * 16, 0, 1);
    v16bf bm = *(const v16bf*)(wbase + (size_t)kt * 512 + (size_t)lane * 16);
    v16bf a0 = load_a_frag(abase,  astride, lane, kt * 32);
    v16bf a1 = load_a_frag(abase1, astride, lane, kt * 32);
    acc0 = __builtin_amdgcn_wmma_f32_16x16x32_bf16(false, a0, false, bm,
                                                   (short)0, acc0, false, false);
    acc1 = __builtin_amdgcn_wmma_f32_16x16x32_bf16(false, a1, false, bm,
                                                   (short)0, acc1, false, false);
  }
}

// ---- pack row-major f32 weight (K x N) into WMMA B-fragment tile layout ----
__global__ void pack_b_kernel(const float* __restrict__ W, bf16_t* __restrict__ dst,
                              int K, int N, int Npad) {
  int KT = K >> 5, NT = Npad >> 4;
  int tid = blockIdx.x * blockDim.x + threadIdx.x;
  if (tid >= KT * NT * 32) return;
  int lane = tid & 31;
  int tile = tid >> 5;
  int kt = tile % KT;
  int nt = tile / KT;
  int n = nt * 16 + (lane & 15);
  int kbase = kt * 32 + ((lane & 16) ? 16 : 0);
  bf16_t* o = dst + (size_t)tile * 512 + lane * 16;
  for (int i = 0; i < 16; ++i) {
    float v = (n < N) ? W[(size_t)(kbase + i) * N + n] : 0.0f;
    o[i] = (bf16_t)v;
  }
}

// ---- magmean[b][d] = mean_f |x[b,f,d]| ----
__global__ void magmean_kernel(const float* __restrict__ x, float* __restrict__ mm) {
  int td = blockIdx.x * 256 + threadIdx.x;   // (b,d), 4096 total
  int b = td >> 9, d = td & 511;
  int f0 = blockIdx.y * (FBINS / 32);
  const float* px = x + ((size_t)b * FBINS + f0) * DIM + d;
  float s = 0.f;
  for (int f = 0; f < FBINS / 32; ++f) s += fabsf(px[(size_t)f * DIM]);
  atomicAdd(&mm[td], s * (1.0f / FBINS));
}

// ---- Km[b][n] = magmean[b] @ k_w + k_b ----
__global__ void km_kernel(const float* __restrict__ mm, const float* __restrict__ kw,
                          const float* __restrict__ kb, float* __restrict__ Km) {
  int t = blockIdx.x * 256 + threadIdx.x;    // 4096
  int b = t >> 9, n = t & 511;
  const float* m = mm + b * DIM;
  float s = kb[n];
  for (int k = 0; k < DIM; ++k) s += m[k] * kw[(size_t)k * DIM + n];
  Km[t] = s;
}

// ---- qkm[b,k,h] = sum_hd q_w[k,h*64+hd]*Km[b,h*64+hd], packed as B tiles ----
__global__ void qkm_kernel(const float* __restrict__ qw, const float* __restrict__ qb,
                           const float* __restrict__ Km, bf16_t* __restrict__ pk,
                           float* __restrict__ qbkm) {
  int t = blockIdx.x * 256 + threadIdx.x;    // 8*512*16
  int n = t & 15, k = (t >> 4) & 511, b = t >> 13;
  float s = 0.f;
  if (n < NHEADS) {
    const float* km = Km + b * DIM + n * HEAD_DIM;
    const float* q  = qw + (size_t)k * DIM + n * HEAD_DIM;
    for (int i = 0; i < HEAD_DIM; ++i) s += q[i] * km[i];
  }
  int kt = k >> 5, kk = k & 31;
  int lane = (kk < 16) ? n : (16 + n);
  pk[((size_t)(b * 16 + kt)) * 512 + lane * 16 + (kk & 15)] = (bf16_t)s;
  if (k == 0 && n < NHEADS) {
    float sb = 0.f;
    const float* km = Km + b * DIM + n * HEAD_DIM;
    for (int i = 0; i < HEAD_DIM; ++i) sb += qb[n * HEAD_DIM + i] * km[i];
    qbkm[b * NHEADS + n] = sb;
  }
}

// ---- main fused kernel: 32 rows per workgroup (two M-tiles), 8 waves ----
__launch_bounds__(256, 1)
__global__ void afs_main_kernel(
    const float* __restrict__ x,
    const float* __restrict__ ln_g, const float* __restrict__ ln_b,
    const float* __restrict__ imp_b1, const float* __restrict__ imp_b2,
    const float* __restrict__ imp_b3,
    const float* __restrict__ b0b1, const float* __restrict__ b1b1,
    const float* __restrict__ b2b1, const float* __restrict__ b3b1,
    const float* __restrict__ b0b2, const float* __restrict__ b1b2,
    const float* __restrict__ b2b2, const float* __restrict__ b3b2,
    const float* __restrict__ spat, const float* __restrict__ pw,
    const bf16_t* __restrict__ pk, const float* __restrict__ qbkm,
    float* __restrict__ finals) {

  __shared__ bf16_t s_mag[MROWS][DIM + 8];
  __shared__ bf16_t s_xn [MROWS][DIM + 8];
  __shared__ bf16_t s_h1 [MROWS][1024 + 8];
  __shared__ bf16_t s_h2 [MROWS][DIM + 8];
  __shared__ float  s_comb[MROWS][16];
  __shared__ float  s_attn[MROWS][16];
  bf16_t* s_bh = &s_h1[0][0];              // alias, stride 968 (960 cols + pad)

  const int row0 = blockIdx.x * MROWS;     // global (b*F+f) row
  const int b    = row0 >> 13;             // / FBINS ; all 32 rows share batch
  const int wave = threadIdx.x >> 5;
  const int lane = threadIdx.x & 31;

  { int r = threadIdx.x >> 4, c = threadIdx.x & 15;
    s_comb[r][c] = 0.f; s_attn[r][c] = 0.f;
    s_comb[r + 16][c] = 0.f; s_attn[r + 16][c] = 0.f; }

  // ---- Phase A: |x|, LayerNorm, bf16 stage into LDS (4 rows / wave) ----
  for (int rr = 0; rr < 4; ++rr) {
    int r = wave * 4 + rr;
    const float* px = x + (size_t)(row0 + r) * DIM + lane * 16;
    float m[16]; float s = 0.f, s2 = 0.f;
    for (int i = 0; i < 16; i += 4) {
      v4f v = *(const v4f*)(px + i);
      for (int j = 0; j < 4; ++j) {
        float a = fabsf(v[j]); m[i + j] = a; s += a; s2 += a * a;
      }
    }
    for (int off = 16; off > 0; off >>= 1) {
      s += __shfl_xor(s, off); s2 += __shfl_xor(s2, off);
    }
    float mu  = s  * (1.0f / DIM);
    float var = s2 * (1.0f / DIM) - mu * mu;
    float inv = rsqrtf(var + 1e-5f);
    for (int i = 0; i < 16; ++i) {
      int d = lane * 16 + i;
      s_mag[r][d] = (bf16_t)m[i];
      s_xn [r][d] = (bf16_t)((m[i] - mu) * inv * ln_g[d] + ln_b[d]);
    }
  }
  __syncthreads();

  // ---- Phase B: h1 = gelu(xn @ imp_w1 + b1)  (64 N-tiles, K=512) ----
  for (int nt = wave; nt < 64; nt += 8) {
    const bf16_t* wp = pk + PK_IMP_W1 + (size_t)nt * 16 * 512;
    v8f a0 = {0.f,0.f,0.f,0.f,0.f,0.f,0.f,0.f}, a1 = a0;
    gemm_tile2(&s_xn[0][0], DIM + 8, wp, 16, lane, a0, a1);
    int n = nt * 16 + (lane & 15);
    float bias = imp_b1[n];
    int mbase = (lane < 16) ? 0 : 8;
    for (int v = 0; v < 8; ++v) {
      s_h1[mbase + v][n]      = (bf16_t)gelu_exact(a0[v] + bias);
      s_h1[16 + mbase + v][n] = (bf16_t)gelu_exact(a1[v] + bias);
    }
  }
  __syncthreads();

  // ---- Phase C: h2 = gelu(h1 @ imp_w2 + b2)  (32 N-tiles, K=1024) ----
  for (int nt = wave; nt < 32; nt += 8) {
    const bf16_t* wp = pk + PK_IMP_W2 + (size_t)nt * 32 * 512;
    v8f a0 = {0.f,0.f,0.f,0.f,0.f,0.f,0.f,0.f}, a1 = a0;
    gemm_tile2(&s_h1[0][0], 1024 + 8, wp, 32, lane, a0, a1);
    int n = nt * 16 + (lane & 15);
    float bias = imp_b2[n];
    int mbase = (lane < 16) ? 0 : 8;
    for (int v = 0; v < 8; ++v) {
      s_h2[mbase + v][n]      = (bf16_t)gelu_exact(a0[v] + bias);
      s_h2[16 + mbase + v][n] = (bf16_t)gelu_exact(a1[v] + bias);
    }
  }
  __syncthreads();

  // ---- Phase D1: band hidden = gelu(mag @ band_w1 + b1)  (60 N-tiles) ----
  const float* bb1[4] = {b0b1, b1b1, b2b1, b3b1};
  for (int t = wave; t < 60; t += 8) {
    int band, ntl, coff; size_t pkoff;
    if      (t < 32) { band = 0; ntl = t;      coff = 0;   pkoff = PK_B0_W1; }
    else if (t < 48) { band = 1; ntl = t - 32; coff = 512; pkoff = PK_B1_W1; }
    else if (t < 56) { band = 2; ntl = t - 48; coff = 768; pkoff = PK_B2_W1; }
    else             { band = 3; ntl = t - 56; coff = 896; pkoff = PK_B3_W1; }
    const bf16_t* wp = pk + pkoff + (size_t)ntl * 16 * 512;
    v8f a0 = {0.f,0.f,0.f,0.f,0.f,0.f,0.f,0.f}, a1 = a0;
    gemm_tile2(&s_mag[0][0], DIM + 8, wp, 16, lane, a0, a1);
    int nloc = ntl * 16 + (lane & 15);
    float bias = bb1[band][nloc];
    int mbase = (lane < 16) ? 0 : 8;
    for (int v = 0; v < 8; ++v) {
      s_bh[(size_t)(mbase + v) * 968 + coff + nloc]      = (bf16_t)gelu_exact(a0[v] + bias);
      s_bh[(size_t)(16 + mbase + v) * 968 + coff + nloc] = (bf16_t)gelu_exact(a1[v] + bias);
    }
  }
  __syncthreads();

  // ---- Phase D2: six 32x16 output GEMMs (neural, 4 bands, attn) ----
  const float* bb2[4] = {b0b2, b1b2, b2b2, b3b2};
  if (wave < 6) {
    const bf16_t* abase; int astride, KT; const bf16_t* wp;
    const float* biasp = nullptr; float scale; bool isAttn = false;
    switch (wave) {
      case 0: abase = &s_h2[0][0]; astride = DIM + 8; KT = 16;
              wp = pk + PK_IMP_W3; biasp = imp_b3; scale = 1.0f;    break;
      case 1: abase = s_bh + 0;    astride = 968;     KT = 16;
              wp = pk + PK_B0_W2;  biasp = bb2[0]; scale = 0.5f;    break;
      case 2: abase = s_bh + 512;  astride = 968;     KT = 8;
              wp = pk + PK_B1_W2;  biasp = bb2[1]; scale = 0.25f;   break;
      case 3: abase = s_bh + 768;  astride = 968;     KT = 4;
              wp = pk + PK_B2_W2;  biasp = bb2[2]; scale = 0.125f;  break;
      case 4: abase = s_bh + 896;  astride = 968;     KT = 2;
              wp = pk + PK_B3_W2;  biasp = bb2[3]; scale = 0.0625f; break;
      default: abase = &s_mag[0][0]; astride = DIM + 8; KT = 16;
              wp = pk + PK_QKM + (size_t)b * 16 * 512;
              scale = 0.3f / 8.0f; isAttn = true;                   break;
    }
    v8f a0 = {0.f,0.f,0.f,0.f,0.f,0.f,0.f,0.f}, a1 = a0;
    gemm_tile2(abase, astride, wp, KT, lane, a0, a1);
    int n = lane & 15;
    if (n < NHEADS) {
      float bias = isAttn ? qbkm[b * NHEADS + n] : biasp[n];
      int mbase = (lane < 16) ? 0 : 8;
      for (int v = 0; v < 8; ++v) {
        float v0 = (a0[v] + bias) * scale;
        float v1 = (a1[v] + bias) * scale;
        if (isAttn) {
          s_attn[mbase + v][n] = v0;
          s_attn[16 + mbase + v][n] = v1;
        } else {
          atomicAdd(&s_comb[mbase + v][n], v0);        // ds_add_f32
          atomicAdd(&s_comb[16 + mbase + v][n], v1);
        }
      }
    }
  }
  __syncthreads();

  // ---- Phase E: pattern modulation + final score (32 rows x 8 heads) ----
  {
    int r = threadIdx.x >> 3, h = threadIdx.x & 7;   // 256 threads = 32x8
    int f = (row0 + r) & (FBINS - 1);
    float t = (float)f * (1.0f / (FBINS - 1));
    float ssum = 0.f;
    for (int j = 0; j < 16; ++j)
      ssum += spat[h * 16 + j] * sinf(6.28318530717958647f * (float)(j + 1) * t);
    float patt = 1.0f / (1.0f + expf(-(ssum * pw[h])));
    float val = s_comb[r][h] * (0.5f + 0.5f * patt) + s_attn[r][h];
    finals[((size_t)(b * NHEADS + h)) * FBINS + f] = val;
  }
}

// ---- per-(b,h) top-k over F via threshold bisection ----
__global__ void topk_kernel(const float* __restrict__ finals,
                            const float* __restrict__ spoff,
                            float* __restrict__ mask_out) {
  int bh = blockIdx.x, b = bh >> 3, h = bh & 7;
  const float* s = finals + (size_t)bh * FBINS;
  float sph = 1.0f / (1.0f + expf(-spoff[h])) * 0.3f + 0.15f;
  int k = (int)((float)FBINS * sph);
  if (k < 1) k = 1; if (k > FBINS) k = FBINS;

  __shared__ float red[256];
  __shared__ int   redi[256];
  __shared__ int   rem;

  float lo = 1e30f, hi = -1e30f;
  for (int i = threadIdx.x; i < FBINS; i += 256) {
    float v = s[i]; lo = fminf(lo, v); hi = fmaxf(hi, v);
  }
  red[threadIdx.x] = lo; __syncthreads();
  for (int st = 128; st > 0; st >>= 1) {
    if (threadIdx.x < st) red[threadIdx.x] = fminf(red[threadIdx.x], red[threadIdx.x + st]);
    __syncthreads();
  }
  lo = red[0]; __syncthreads();
  red[threadIdx.x] = hi; __syncthreads();
  for (int st = 128; st > 0; st >>= 1) {
    if (threadIdx.x < st) red[threadIdx.x] = fmaxf(red[threadIdx.x], red[threadIdx.x + st]);
    __syncthreads();
  }
  hi = red[0]; __syncthreads();

  float a = lo - (fabsf(lo) * 1e-4f + 1e-6f), c = hi;
  for (int it = 0; it < 32; ++it) {
    float mid = 0.5f * (a + c);
    int cnt = 0;
    for (int i = threadIdx.x; i < FBINS; i += 256) cnt += (s[i] > mid) ? 1 : 0;
    redi[threadIdx.x] = cnt; __syncthreads();
    for (int st = 128; st > 0; st >>= 1) {
      if (threadIdx.x < st) redi[threadIdx.x] += redi[threadIdx.x + st];
      __syncthreads();
    }
    cnt = redi[0]; __syncthreads();
    if (cnt > k) a = mid; else c = mid;
  }
  int cg = 0;
  for (int i = threadIdx.x; i < FBINS; i += 256) cg += (s[i] > c) ? 1 : 0;
  redi[threadIdx.x] = cg; __syncthreads();
  for (int st = 128; st > 0; st >>= 1) {
    if (threadIdx.x < st) redi[threadIdx.x] += redi[threadIdx.x + st];
    __syncthreads();
  }
  if (threadIdx.x == 0) rem = k - redi[0];
  __syncthreads();

  for (int i = threadIdx.x; i < FBINS; i += 256) {
    float v = s[i], mv = 0.f;
    if (v > c) mv = 1.f;
    else if (v > a && rem > 0) { if (atomicSub(&rem, 1) > 0) mv = 1.f; }
    mask_out[((size_t)b * FBINS + i) * NHEADS + h] = mv;
  }
}

// ---- x_filtered = x * mask (broadcast over head_dim) ----
__global__ void filter_kernel(const float* __restrict__ x,
                              const float* __restrict__ mask,
                              float* __restrict__ out) {
  size_t i = (size_t)blockIdx.x * 256 + threadIdx.x;
  int d = (int)(i & 511);
  size_t bf = i >> 9;
  out[i] = x[i] * mask[bf * NHEADS + (d >> 6)];
}

extern "C" void kernel_launch(void* const* d_in, const int* in_sizes, int n_in,
                              void* d_out, int out_size, void* d_ws, size_t ws_size,
                              hipStream_t stream) {
  const float* x      = (const float*)d_in[0];
  const float* ln_g   = (const float*)d_in[1];
  const float* ln_b   = (const float*)d_in[2];
  const float* imp_w1 = (const float*)d_in[3];
  const float* imp_b1 = (const float*)d_in[4];
  const float* imp_w2 = (const float*)d_in[5];
  const float* imp_b2 = (const float*)d_in[6];
  const float* imp_w3 = (const float*)d_in[7];
  const float* imp_b3 = (const float*)d_in[8];
  const float* bw1[4] = {(const float*)d_in[9],  (const float*)d_in[13],
                         (const float*)d_in[17], (const float*)d_in[21]};
  const float* bb1[4] = {(const float*)d_in[10], (const float*)d_in[14],
                         (const float*)d_in[18], (const float*)d_in[22]};
  const float* bw2[4] = {(const float*)d_in[11], (const float*)d_in[15],
                         (const float*)d_in[19], (const float*)d_in[23]};
  const float* bb2[4] = {(const float*)d_in[12], (const float*)d_in[16],
                         (const float*)d_in[20], (const float*)d_in[24]};
  const float* spat   = (const float*)d_in[25];
  const float* pw     = (const float*)d_in[26];
  const float* spoff  = (const float*)d_in[28];   // selection_temp (27) unused
  const float* q_w    = (const float*)d_in[29];
  const float* q_b    = (const float*)d_in[30];
  const float* k_w    = (const float*)d_in[31];
  const float* k_b    = (const float*)d_in[32];

  char* ws = (char*)d_ws;
  bf16_t* pk     = (bf16_t*)ws;
  float* magmean = (float*)(ws + WS_MAGMEAN);
  float* Km      = (float*)(ws + WS_KM);
  float* qbkm    = (float*)(ws + WS_QBKM);
  float* finals  = (float*)(ws + WS_FINAL);

  float* outX    = (float*)d_out;
  float* outMask = outX + (size_t)BATCH * FBINS * DIM;

  auto packLaunch = [&](const float* W, size_t pkoff, int K, int N, int Npad) {
    int threads = (K >> 5) * (Npad >> 4) * 32;
    pack_b_kernel<<<(threads + 255) / 256, 256, 0, stream>>>(W, pk + pkoff, K, N, Npad);
  };
  packLaunch(imp_w1, PK_IMP_W1, 512, 1024, 1024);
  packLaunch(imp_w2, PK_IMP_W2, 1024, 512, 512);
  packLaunch(imp_w3, PK_IMP_W3, 512, 8, 16);
  packLaunch(bw1[0], PK_B0_W1, 512, 512, 512);
  packLaunch(bw1[1], PK_B1_W1, 512, 256, 256);
  packLaunch(bw1[2], PK_B2_W1, 512, 128, 128);
  packLaunch(bw1[3], PK_B3_W1, 512, 64, 64);
  packLaunch(bw2[0], PK_B0_W2, 512, 8, 16);
  packLaunch(bw2[1], PK_B1_W2, 256, 8, 16);
  packLaunch(bw2[2], PK_B2_W2, 128, 8, 16);
  packLaunch(bw2[3], PK_B3_W2, 64, 8, 16);

  hipMemsetAsync(magmean, 0, BATCH * DIM * sizeof(float), stream);
  magmean_kernel<<<dim3(BATCH * DIM / 256, 32), 256, 0, stream>>>(x, magmean);
  km_kernel<<<BATCH * DIM / 256, 256, 0, stream>>>(magmean, k_w, k_b, Km);
  qkm_kernel<<<BATCH * DIM * 16 / 256, 256, 0, stream>>>(q_w, q_b, Km, pk + PK_QKM, qbkm);

  afs_main_kernel<<<(BATCH * FBINS) / MROWS, 256, 0, stream>>>(
      x, ln_g, ln_b, imp_b1, imp_b2, imp_b3,
      bb1[0], bb1[1], bb1[2], bb1[3], bb2[0], bb2[1], bb2[2], bb2[3],
      spat, pw, pk, qbkm, finals);

  topk_kernel<<<BATCH * NHEADS, 256, 0, stream>>>(finals, spoff, outMask);

  filter_kernel<<<((size_t)BATCH * FBINS * DIM) / 256, 256, 0, stream>>>(x, outMask, outX);

  (void)in_sizes; (void)n_in; (void)out_size; (void)ws_size;
}
